// TAISP_76355928588665
// MI455X (gfx1250) — matrix operand order repaired and checked by reading, not compiled, not run on hardware
//
#include <hip/hip_runtime.h>
#include <hip/hip_bf16.h>

typedef __attribute__((ext_vector_type(16))) _Float16 v16h;
typedef __attribute__((ext_vector_type(8)))  _Float16 v8h;
typedef __attribute__((ext_vector_type(8)))  float    v8f;

union AFrag { v16h v; v8h h[2]; };

static __device__ __forceinline__ int imin(int a, int b) { return a < b ? a : b; }
static __device__ __forceinline__ float clamp01(float v) { return fminf(fmaxf(v, 1e-6f), 1.0f); }

// ---------------- workspace layout (bytes, 256-aligned) ----------------
static constexpr size_t OFF_RZ    = 0;            // 4*256*256*3 f32   = 3,145,728
static constexpr size_t OFF_C1    = 3145728;      // 4*128*128*32 f16  = 4,194,304
static constexpr size_t OFF_C2    = 7340032;      // 4*64*64*32 f16    = 1,048,576
static constexpr size_t OFF_C3    = 8388608;      // 4*32*32*32 f32    =   524,288
static constexpr size_t OFF_PB2   = 8912896;      // 9*2*32*16 f16     =    18,432
static constexpr size_t OFF_PB3   = 8931328;      // 9*2*32*16 f16     =    18,432
static constexpr size_t OFF_G     = 8949760;      // 4*32 f32
static constexpr size_t OFF_DG    = 8950272;      // 4 f32
static constexpr size_t OFF_CCM   = 8950528;      // 4*27 f32
static constexpr size_t OFF_GAM   = 8951040;      // 4*75 f32
static constexpr size_t OFF_WB    = 8952320;      // 2 f32 (gr, gb)
static constexpr size_t OFF_PART  = 8952576;      // 16384*3 f32 = 196,608
static constexpr size_t OFF_INTER = 9149184;      // 4*3*1024*1024 f32 = 50,331,648
// total ws need: 59,480,832 bytes

// ---------------- stage 1: bilinear 1024 -> 256 (channels-last out) ----------------
__global__ void kResize(const float* __restrict__ img, float* __restrict__ rz) {
  int t = blockIdx.x * 256 + threadIdx.x;        // 4*256*256 threads
  int b = t >> 16; int r = t & 65535; int y = r >> 8; int x = r & 255;
  float fy = y * (1023.0f / 255.0f);
  float fx = x * (1023.0f / 255.0f);
  int y0 = (int)fy; int y1 = imin(y0 + 1, 1023); float wy = fy - (float)y0;
  int x0 = (int)fx; int x1 = imin(x0 + 1, 1023); float wx = fx - (float)x0;
#pragma unroll
  for (int c = 0; c < 3; ++c) {
    const float* p = img + (((size_t)(b * 3 + c)) << 20);
    float t0 = p[y0 * 1024 + x0] * (1.0f - wx) + p[y0 * 1024 + x1] * wx;
    float t1 = p[y1 * 1024 + x0] * (1.0f - wx) + p[y1 * 1024 + x1] * wx;
    rz[(size_t)t * 3 + c] = t0 * (1.0f - wy) + t1 * wy;
  }
}

// ---------------- pack conv2/conv3 weights into per-lane WMMA B-fragments ----------------
// B (K=32 x N=16) layout: lane L holds column N = L%16, element e -> K = (e&7) + 8*(L/16) + 16*(e/8)
__global__ void kPackW(const float* __restrict__ w2, const float* __restrict__ w3,
                       _Float16* __restrict__ p2, _Float16* __restrict__ p3) {
  int t = blockIdx.x * blockDim.x + threadIdx.x;   // 2*9*2*32 = 1152 items
  if (t >= 1152) return;
  int conv = t / 576; int r = t % 576;
  int kpos = r / 64;  int r2 = r % 64;
  int half = r2 / 32; int lane = r2 % 32;
  const float* w = conv ? w3 : w2;
  _Float16* dst = (conv ? p3 : p2) + ((size_t)((kpos * 2 + half) * 32 + lane)) * 16;
  int n  = half * 16 + (lane & 15);
  int ky = kpos / 3, kx = kpos % 3;
#pragma unroll
  for (int e = 0; e < 16; ++e) {
    int cin = (e & 7) + 8 * (lane >> 4) + 16 * (e >> 3);
    dst[e] = (_Float16)w[((n * 32 + cin) * 3 + ky) * 3 + kx];
  }
}

// ---------------- conv1: 3->32, stride 2, pad 1, ReLU (direct, small K) ----------------
__global__ void kConv1(const float* __restrict__ rz, const float* __restrict__ w,
                       const float* __restrict__ bias, _Float16* __restrict__ out) {
  __shared__ float ws[32 * 27];
  __shared__ float bs[32];
  for (int i = threadIdx.x; i < 32 * 27; i += blockDim.x) ws[i] = w[i];
  if (threadIdx.x < 32) bs[threadIdx.x] = bias[threadIdx.x];
  __syncthreads();
  int b = blockIdx.y;
  int pix = blockIdx.x * 256 + threadIdx.x;       // 128*128 per batch
  int oy = pix >> 7; int ox = pix & 127;
  float acc[32];
#pragma unroll
  for (int c = 0; c < 32; ++c) acc[c] = bs[c];
#pragma unroll
  for (int ky = 0; ky < 3; ++ky) {
    int iy = 2 * oy - 1 + ky;
    if (iy < 0 || iy >= 256) continue;
#pragma unroll
    for (int kx = 0; kx < 3; ++kx) {
      int ix = 2 * ox - 1 + kx;
      if (ix < 0 || ix >= 256) continue;
      const float* pv = rz + (((size_t)b * 256 + iy) * 256 + ix) * 3;
      float v0 = pv[0], v1 = pv[1], v2 = pv[2];
      int kb = ky * 3 + kx;
#pragma unroll
      for (int co = 0; co < 32; ++co) {
        const float* wc = ws + co * 27 + kb;      // [co][ci][kpos] = co*27 + ci*9 + kpos
        acc[co] += v0 * wc[0] + v1 * wc[9] + v2 * wc[18];
      }
    }
  }
  _Float16* po = out + (((size_t)b * 128 + oy) * 128 + ox) * 32;
#pragma unroll
  for (int co = 0; co < 32; ++co) po[co] = (_Float16)(acc[co] > 0.0f ? acc[co] : 0.0f);
}

// ---------------- conv2 / conv3: implicit GEMM via v_wmma_f32_16x16x32_f16 ----------------
// in: [B][Hin][Win][32] f16 channels-last; one wave computes 16 output pixels x 32 cout
template <bool STORE_F32>
__global__ void kConvWMMA(const _Float16* __restrict__ in, const _Float16* __restrict__ packB,
                          const float* __restrict__ bias, _Float16* __restrict__ outH,
                          float* __restrict__ outF, int Hin, int Win, int Hout, int Wout) {
  const int lane = threadIdx.x;                  // 0..31 (wave32, EXEC all ones)
  const int tilesX = Wout >> 4;
  const int oy  = blockIdx.x / tilesX;
  const int ox0 = (blockIdx.x % tilesX) << 4;
  const int b   = blockIdx.y;
  const int m   = lane & 15;                     // A-fragment row M
  const int hi  = lane >> 4;                     // lane half selects cin sub-blocks
  const int ox  = ox0 + m;

  v8f acc0 = {}; v8f acc1 = {};
  const _Float16* inB = in + (size_t)b * Hin * Win * 32;
  const v8h zero = {};

#pragma unroll
  for (int ky = 0; ky < 3; ++ky) {
    int iy = 2 * oy - 1 + ky;
    bool yok = (iy >= 0) && (iy < Hin);
#pragma unroll
    for (int kx = 0; kx < 3; ++kx) {
      int ix = 2 * ox - 1 + kx;
      AFrag a;
      if (yok && ix >= 0 && ix < Win) {
        // lane<16 reads cin {0..7,16..23}; lane>=16 reads {8..15,24..31}
        const _Float16* p = inB + ((size_t)iy * Win + ix) * 32 + hi * 8;
        a.h[0] = *(const v8h*)(p);
        a.h[1] = *(const v8h*)(p + 16);
      } else {
        a.h[0] = zero; a.h[1] = zero;
      }
      int kpos = ky * 3 + kx;
      v16h b0 = *(const v16h*)(packB + ((size_t)((kpos * 2 + 0) * 32 + lane)) * 16);
      v16h b1 = *(const v16h*)(packB + ((size_t)((kpos * 2 + 1) * 32 + lane)) * 16);
      acc0 = __builtin_amdgcn_wmma_f32_16x16x32_f16(false, a.v, false, b0, (short)0, acc0, false, false);
      acc1 = __builtin_amdgcn_wmma_f32_16x16x32_f16(false, a.v, false, b1, (short)0, acc1, false, false);
    }
  }
  // D layout: lanes 0-15 hold M=v (VGPR v), lanes 16-31 hold M=8+v; N = lane&15
  int n = lane & 15;
  float bia0 = bias[n], bia1 = bias[n + 16];
#pragma unroll
  for (int v = 0; v < 8; ++v) {
    int mm = v + 8 * hi;
    size_t pix = ((size_t)b * Hout + oy) * Wout + (ox0 + mm);
    float v0 = acc0[v] + bia0; v0 = v0 > 0.0f ? v0 : 0.0f;
    float v1 = acc1[v] + bia1; v1 = v1 > 0.0f ? v1 : 0.0f;
    if (STORE_F32) { outF[pix * 32 + n] = v0; outF[pix * 32 + n + 16] = v1; }
    else           { outH[pix * 32 + n] = (_Float16)v0; outH[pix * 32 + n + 16] = (_Float16)v1; }
  }
}

// ---------------- g = spatial mean of conv3 output ----------------
__global__ void kReduceG(const float* __restrict__ c3, float* __restrict__ g) {
  int b = blockIdx.x; int t = threadIdx.x;
  int ch = t & 31; int grp = t >> 5;             // 4 groups x 32 channels
  const float* p = c3 + (size_t)b * 1024 * 32;
  float s = 0.0f;
  for (int px = grp; px < 1024; px += 4) s += p[px * 32 + ch];
  __shared__ float sm[128];
  sm[t] = s; __syncthreads();
  if (grp == 0) g[b * 32 + ch] = (sm[ch] + sm[32 + ch] + sm[64 + ch] + sm[96 + ch]) * (1.0f / 1024.0f);
}

// ---------------- tiny GEMVs: dgain / CCM / gamma parameters ----------------
__global__ void kParams(const float* __restrict__ g,
                        const float* __restrict__ gamma_w, const float* __restrict__ gamma_b,
                        const float* __restrict__ dgain_w, const float* __restrict__ dgain_b,
                        const float* __restrict__ ccm_w, const float* __restrict__ ccm_b,
                        float* __restrict__ dgain, float* __restrict__ ccm, float* __restrict__ gamma) {
  int t = blockIdx.x * blockDim.x + threadIdx.x;
  if (t >= 4 * 103) return;
  int b = t / 103; int r = t % 103;
  const float* gv = g + b * 32;
  const float* W; const float* Bv; int j;
  if (r == 0)      { W = dgain_w; Bv = dgain_b; j = 0; }
  else if (r < 28) { W = ccm_w;   Bv = ccm_b;   j = r - 1; }
  else             { W = gamma_w; Bv = gamma_b; j = r - 28; }
  float s = Bv[j];
#pragma unroll
  for (int k = 0; k < 32; ++k) s += gv[k] * W[j * 32 + k];
  if (r == 0)      dgain[b]          = 1.0f + 0.1f / (1.0f + __expf(-s));
  else if (r < 28) ccm[b * 27 + j]   = 2.0f * s + 1.0f;
  else             gamma[b * 75 + j] = 2.2f + 0.8f / (1.0f + __expf(-s));
}

// ---------------- full-res pass 1: bilateral + denoise blend + dgain + channel partial sums ----------------
__global__ void kBilateral(const float* __restrict__ img, const float* __restrict__ dgain,
                           float* __restrict__ inter, float* __restrict__ partials) {
  int b = blockIdx.y;
  int pix = blockIdx.x * 256 + threadIdx.x;      // H*W per batch
  int y = pix >> 10; int x = pix & 1023;
  const float e1 = 0.60653066f;                  // exp(-0.5)
  const float e2 = 0.36787944f;                  // exp(-1.0)
  const float ksum = 1.0f + 4.0f * e1 + 4.0f * e2;
  const float ks[9] = { e2/ksum, e1/ksum, e2/ksum, e1/ksum, 1.0f/ksum, e1/ksum, e2/ksum, e1/ksum, e2/ksum };
  float dg = dgain[b];
  int ym = (y == 0)    ? 1    : y - 1;           // jnp.pad mode='reflect'
  int yp = (y == 1023) ? 1022 : y + 1;
  int xm = (x == 0)    ? 1    : x - 1;
  int xp = (x == 1023) ? 1022 : x + 1;
  int ys3[3] = { ym, y, yp };
  int xs3[3] = { xm, x, xp };
  // hint the other channel planes into cache (gfx1250 global_prefetch_b8)
  __builtin_prefetch(img + (((size_t)(b * 3 + 1)) << 20) + pix, 0, 1);
  __builtin_prefetch(img + (((size_t)(b * 3 + 2)) << 20) + pix, 0, 1);
  float csum[3];
#pragma unroll
  for (int c = 0; c < 3; ++c) {
    const float* p = img + (((size_t)(b * 3 + c)) << 20);
    float v = p[y * 1024 + x];
    float num = 0.0f, den = 0.0f;
#pragma unroll
    for (int i = 0; i < 3; ++i)
#pragma unroll
      for (int jj = 0; jj < 3; ++jj) {
        float q = p[ys3[i] * 1024 + xs3[jj]];
        float d = q - v;
        float wgt = __expf(-d * d * 200.0f) * ks[i * 3 + jj];  // 1/(2*0.05^2)=200
        num += wgt * q; den += wgt;
      }
    float bf = num / den;
    float de = clamp01(0.5f * (v + bf));
    float it = clamp01(de * dg);
    inter[(((size_t)(b * 3 + c)) << 20) + pix] = it;
    csum[c] = it;
  }
  // deterministic block reduction of channel sums
  __shared__ float sm[256];
  int blk = b * gridDim.x + blockIdx.x;
#pragma unroll
  for (int c = 0; c < 3; ++c) {
    sm[threadIdx.x] = csum[c]; __syncthreads();
    for (int s = 128; s > 0; s >>= 1) {
      if (threadIdx.x < s) sm[threadIdx.x] += sm[threadIdx.x + s];
      __syncthreads();
    }
    if (threadIdx.x == 0) partials[(size_t)blk * 3 + c] = sm[0];
    __syncthreads();
  }
}

// ---------------- deterministic final reduce -> WB gains gr=mg/mr, gb=mg/mb ----------------
__global__ void kWB(const float* __restrict__ partials, float* __restrict__ wb) {
  __shared__ float sm[3 * 256];
  float s0 = 0.0f, s1 = 0.0f, s2 = 0.0f;
  for (int i = threadIdx.x; i < 16384; i += 256) {
    s0 += partials[(size_t)i * 3 + 0];
    s1 += partials[(size_t)i * 3 + 1];
    s2 += partials[(size_t)i * 3 + 2];
  }
  sm[threadIdx.x] = s0; sm[256 + threadIdx.x] = s1; sm[512 + threadIdx.x] = s2;
  __syncthreads();
  for (int s = 128; s > 0; s >>= 1) {
    if (threadIdx.x < s) {
      sm[threadIdx.x]       += sm[threadIdx.x + s];
      sm[256 + threadIdx.x] += sm[256 + threadIdx.x + s];
      sm[512 + threadIdx.x] += sm[512 + threadIdx.x + s];
    }
    __syncthreads();
  }
  if (threadIdx.x == 0) { wb[0] = sm[256] / sm[0]; wb[1] = sm[256] / sm[512]; }
}

// ---------------- full-res pass 2: WB + polynomial CCM + per-pixel gamma + output ----------------
__global__ void kFinal(const float* __restrict__ inter, const float* __restrict__ ccm,
                       const float* __restrict__ gamma, const float* __restrict__ wb,
                       float* __restrict__ out) {
  int b = blockIdx.y;
  int pix = blockIdx.x * 256 + threadIdx.x;
  __shared__ float sc[27]; __shared__ float sg[75]; __shared__ float swb[2];
  if (threadIdx.x < 27) sc[threadIdx.x] = ccm[b * 27 + threadIdx.x];
  if (threadIdx.x >= 32 && threadIdx.x < 107) sg[threadIdx.x - 32] = gamma[b * 75 + threadIdx.x - 32];
  if (threadIdx.x >= 128 && threadIdx.x < 130) swb[threadIdx.x - 128] = wb[threadIdx.x - 128];
  __syncthreads();
  int y = pix >> 10, x = pix & 1023;
  size_t base = ((size_t)b * 3) << 20;
  float R  = clamp01(inter[base + pix] * swb[0]);
  float G  = clamp01(inter[base + (1 << 20) + pix]);
  float Bc = clamp01(inter[base + (2 << 20) + pix] * swb[1]);
  float X[9] = { R, G, Bc, R * R, G * G, Bc * Bc, R * G, R * Bc, G * Bc };
  // bilinear upsample of 5x5 gamma map
  float fy = y * (4.0f / 1023.0f), fx = x * (4.0f / 1023.0f);
  int y0 = (int)fy; int y1 = imin(y0 + 1, 4); float wy = fy - (float)y0;
  int x0 = (int)fx; int x1 = imin(x0 + 1, 4); float wx = fx - (float)x0;
  float invg[3];
#pragma unroll
  for (int j = 0; j < 3; ++j) {
    const float* gp = sg + j * 25;
    float t0 = gp[y0 * 5 + x0] * (1.0f - wx) + gp[y0 * 5 + x1] * wx;
    float t1 = gp[y1 * 5 + x0] * (1.0f - wx) + gp[y1 * 5 + x1] * wx;
    invg[j] = 1.0f / (t0 * (1.0f - wy) + t1 * wy);
  }
#pragma unroll
  for (int c = 0; c < 3; ++c) {
    float s = 0.0f;
#pragma unroll
    for (int j = 0; j < 9; ++j) s += sc[c * 9 + j] * X[j];
    s = clamp01(s);
    float l = __logf(s);
    float o = (__expf(l * invg[0]) + __expf(l * invg[1]) + __expf(l * invg[2])) * (1.0f / 3.0f);
    out[base + ((size_t)c << 20) + pix] = o;
  }
}

extern "C" void kernel_launch(void* const* d_in, const int* in_sizes, int n_in,
                              void* d_out, int out_size, void* d_ws, size_t ws_size,
                              hipStream_t stream) {
  const float* img     = (const float*)d_in[0];
  const float* conv1_w = (const float*)d_in[1];
  const float* conv1_b = (const float*)d_in[2];
  const float* conv2_w = (const float*)d_in[3];
  const float* conv2_b = (const float*)d_in[4];
  const float* conv3_w = (const float*)d_in[5];
  const float* conv3_b = (const float*)d_in[6];
  const float* gamma_w = (const float*)d_in[7];
  const float* gamma_b = (const float*)d_in[8];
  const float* dgain_w = (const float*)d_in[9];
  const float* dgain_b = (const float*)d_in[10];
  const float* ccm_w   = (const float*)d_in[11];
  const float* ccm_b   = (const float*)d_in[12];

  char* ws = (char*)d_ws;
  float*    rz    = (float*)(ws + OFF_RZ);
  _Float16* c1    = (_Float16*)(ws + OFF_C1);
  _Float16* c2    = (_Float16*)(ws + OFF_C2);
  float*    c3    = (float*)(ws + OFF_C3);
  _Float16* pb2   = (_Float16*)(ws + OFF_PB2);
  _Float16* pb3   = (_Float16*)(ws + OFF_PB3);
  float*    g     = (float*)(ws + OFF_G);
  float*    dgA   = (float*)(ws + OFF_DG);
  float*    ccmA  = (float*)(ws + OFF_CCM);
  float*    gamA  = (float*)(ws + OFF_GAM);
  float*    wbA   = (float*)(ws + OFF_WB);
  float*    part  = (float*)(ws + OFF_PART);
  float*    inter = (float*)(ws + OFF_INTER);

  // stats path
  kResize<<<dim3(1024), dim3(256), 0, stream>>>(img, rz);
  kPackW<<<dim3(5), dim3(256), 0, stream>>>(conv2_w, conv3_w, pb2, pb3);
  kConv1<<<dim3(64, 4), dim3(256), 0, stream>>>(rz, conv1_w, conv1_b, c1);
  kConvWMMA<false><<<dim3(256, 4), dim3(32), 0, stream>>>(c1, pb2, conv2_b, c2, nullptr, 128, 128, 64, 64);
  kConvWMMA<true ><<<dim3(64, 4),  dim3(32), 0, stream>>>(c2, pb3, conv3_b, nullptr, c3, 64, 64, 32, 32);
  kReduceG<<<dim3(4), dim3(128), 0, stream>>>(c3, g);
  kParams<<<dim3(2), dim3(256), 0, stream>>>(g, gamma_w, gamma_b, dgain_w, dgain_b, ccm_w, ccm_b,
                                             dgA, ccmA, gamA);
  // full-res path (memory-bound: two fused passes)
  kBilateral<<<dim3(4096, 4), dim3(256), 0, stream>>>(img, dgA, inter, part);
  kWB<<<dim3(1), dim3(256), 0, stream>>>(part, wbA);
  kFinal<<<dim3(4096, 4), dim3(256), 0, stream>>>(inter, ccmA, gamA, wbA, (float*)d_out);
}
